// Pyramid_OC_Module_61890478735835
// MI455X (gfx1250) — matrix operand order repaired and checked by reading, not compiled, not run on hardware
//
#include <hip/hip_runtime.h>
#include <hip/hip_bf16.h>
#include <cstdint>
#include <cstddef>

typedef __attribute__((ext_vector_type(16))) __bf16 v16bf;
typedef __attribute__((ext_vector_type(8)))  __bf16 v8bf;
typedef __attribute__((ext_vector_type(8)))  float  v8f;

#define B_    4
#define H_    192
#define W_    192
#define C_    256
#define PIXPB (H_*W_)        // 36864 pixels per image
#define NPIX  (B_*PIXPB)     // 147456
#define NTOK  50             // 1 + 4 + 9 + 36 tokens per image
#define EPS_  1e-3f

struct AttnPtrs {
    const float *wf[4], *bf[4], *wg[4], *bg[4], *wh[4], *bh[4], *wo[4], *bo[4], *gamma[4];
};

__device__ __forceinline__ int imin_(int a, int b){ return a < b ? a : b; }
__device__ __forceinline__ int imax_(int a, int b){ return a > b ? a : b; }

// ---------------------------------------------------------------- token decode
__device__ __forceinline__ void tok_decode(int t, int& lvl, int& iy, int& ix, int& li) {
    if (t == 0)       { lvl = 1; li = 0; iy = 0; ix = 0; }
    else if (t < 5)   { lvl = 2; li = 1; int i = t - 1;  iy = i >> 1; ix = i & 1; }
    else if (t < 14)  { lvl = 3; li = 2; int i = t - 5;  iy = i / 3;  ix = i % 3; }
    else              { lvl = 6; li = 3; int i = t - 14; iy = i / 6;  ix = i % 6; }
}

// half-pixel bilinear weight of grid point i for sample coord s, clamped edges
__device__ __forceinline__ float edge_w(int i, float s, int lvl) {
    float fl = floorf(s);
    float f  = s - fl;
    int i0 = (int)fl;
    int j0 = imin_(imax_(i0, 0), lvl - 1);
    int j1 = imin_(imax_(i0 + 1, 0), lvl - 1);
    float w = 0.0f;
    if (i == j0) w += 1.0f - f;
    if (i == j1) w += f;
    return w;
}

__device__ __forceinline__ float tokw(int k, int h, int w) {
    if (k >= NTOK) return 0.0f;
    int lvl, iy, ix, li;
    tok_decode(k, lvl, iy, ix, li);
    float scale = (float)lvl * (1.0f / 192.0f);
    float sy = ((float)h + 0.5f) * scale - 0.5f;
    float sx = ((float)w + 0.5f) * scale - 0.5f;
    return edge_w(iy, sy, lvl) * edge_w(ix, sx, lvl);
}

// K index inside a 32-wide K-step for 16-bit WMMA A-fragment element e, lane-half lh
__device__ __forceinline__ int afrag_k(int lh, int e) {
    return (e < 8) ? (lh * 8 + e) : (16 + lh * 8 + (e - 8));
}

// ---------------------------------------------------------------- init (zero Zb pads)
__global__ void k_init(float* __restrict__ zp, int n) {
    int i = blockIdx.x * blockDim.x + threadIdx.x;
    if (i < n) zp[i] = 0.0f;
}

// ---------------------------------------------------------------- partial 32x32 block sums
// grid: B*36*8 blocks, 256 threads; part6[((b*36+s)*8+g)*256+ch] = sum over 4 rows x 32 cols
__global__ __launch_bounds__(256) void k_pool6(const float* __restrict__ x, float* __restrict__ part6) {
    int id = blockIdx.x;
    int b  = id / (36 * 8);
    int r  = id % (36 * 8);
    int s  = r >> 3;
    int g  = r & 7;
    int sy = s / 6, sx = s % 6;
    int ch = threadIdx.x;
    float acc = 0.0f;
    for (int iy = g * 4; iy < g * 4 + 4; ++iy) {
        const float* row = x + (size_t)((b * H_ + sy * 32 + iy) * W_ + sx * 32) * C_ + ch;
        for (int ix = 0; ix < 32; ++ix) acc += row[(size_t)ix * C_];
    }
    part6[(size_t)((b * 36 + s) * 8 + g) * C_ + ch] = acc;
}

// ---------------------------------------------------------------- pooled means for all levels
// grid: B*50 blocks, 256 threads
__global__ __launch_bounds__(256) void k_pool_all(const float* __restrict__ part6, float* __restrict__ pooled) {
    int id = blockIdx.x;
    int b = id / NTOK, t = id % NTOK;
    int lvl, iy, ix, li;
    tok_decode(t, lvl, iy, ix, li);
    int bs = 6 / lvl;          // 32x32 sub-blocks per pooled cell side
    int ch = threadIdx.x;
    float acc = 0.0f;
    for (int dy = 0; dy < bs; ++dy)
        for (int dx = 0; dx < bs; ++dx) {
            int s = (iy * bs + dy) * 6 + (ix * bs + dx);
            const float* p = part6 + (size_t)((b * 36 + s) * 8) * C_ + ch;
            for (int g = 0; g < 8; ++g) acc += p[(size_t)g * C_];
        }
    pooled[(size_t)(b * NTOK + t) * C_ + ch] = acc / (float)(bs * bs * 1024);
}

// ---------------------------------------------------------------- f,g,h projections
// grid: B*50 blocks, 192 threads; fgh layout per token: [0:32)=f, [32:64)=g, [64:192)=h
__global__ __launch_bounds__(192) void k_qkv(const float* __restrict__ pooled, float* __restrict__ fgh, AttnPtrs ap) {
    int id = blockIdx.x;
    int b = id / NTOK, t = id % NTOK;
    int lvl, iy, ix, li;
    tok_decode(t, lvl, iy, ix, li);
    int j = threadIdx.x;
    const float* pr = pooled + (size_t)(b * NTOK + t) * C_;
    float acc;
    if (j < 32) {
        const float* w = ap.wf[li]; acc = ap.bf[li][j];
        for (int c = 0; c < C_; ++c) acc += pr[c] * w[c * 32 + j];
    } else if (j < 64) {
        int jj = j - 32;
        const float* w = ap.wg[li]; acc = ap.bg[li][jj];
        for (int c = 0; c < C_; ++c) acc += pr[c] * w[c * 32 + jj];
    } else {
        int jj = j - 64;
        const float* w = ap.wh[li]; acc = ap.bh[li][jj];
        for (int c = 0; c < C_; ++c) acc += pr[c] * w[c * 128 + jj];
    }
    fgh[(size_t)(b * NTOK + t) * 192 + j] = acc;
}

// ---------------------------------------------------------------- attention per (b, level)
// grid: 16 blocks, 256 threads
__global__ __launch_bounds__(256) void k_attn(const float* __restrict__ pooled, const float* __restrict__ fgh,
                                              float* __restrict__ attnout, AttnPtrs ap) {
    const int lvls[4] = {1, 2, 3, 6};
    const int toff[4] = {0, 1, 5, 14};
    int b  = blockIdx.x >> 2;
    int li = blockIdx.x & 3;
    int lvl = lvls[li], n = lvl * lvl, tb = toff[li];
    __shared__ float sf[36][32];
    __shared__ float sg[36][32];
    __shared__ float sbeta[36][36];
    __shared__ float so[36][128];
    int tid = threadIdx.x;

    for (int idx = tid; idx < n * 64; idx += 256) {
        int i = idx >> 6, c = idx & 63;
        float v = fgh[(size_t)(b * NTOK + tb + i) * 192 + c];
        if (c < 32) sf[i][c] = v; else sg[i][c - 32] = v;
    }
    __syncthreads();
    for (int idx = tid; idx < n * n; idx += 256) {
        int i = idx / n, m = idx % n;
        float d = 0.0f;
        for (int c = 0; c < 32; ++c) d += sg[i][c] * sf[m][c];
        sbeta[i][m] = d;
    }
    __syncthreads();
    if (tid < n) {
        float mx = -3.0e38f;
        for (int m = 0; m < n; ++m) mx = fmaxf(mx, sbeta[tid][m]);
        float sm = 0.0f;
        for (int m = 0; m < n; ++m) { float e = expf(sbeta[tid][m] - mx); sbeta[tid][m] = e; sm += e; }
        float inv = 1.0f / sm;
        for (int m = 0; m < n; ++m) sbeta[tid][m] *= inv;
    }
    __syncthreads();
    for (int idx = tid; idx < n * 128; idx += 256) {
        int i = idx >> 7, c = idx & 127;
        float d = 0.0f;
        for (int m = 0; m < n; ++m) d += sbeta[i][m] * fgh[(size_t)(b * NTOK + tb + m) * 192 + 64 + c];
        so[i][c] = d;
    }
    __syncthreads();
    float gm = ap.gamma[li][0];
    const float* wo = ap.wo[li];
    const float* bo = ap.bo[li];
    for (int idx = tid; idx < n * 256; idx += 256) {
        int i = idx >> 8, c = idx & 255;
        float d = bo[c];
        for (int k = 0; k < 128; ++k) d += so[i][k] * wo[k * 256 + c];
        size_t o = (size_t)(b * NTOK + tb + i) * C_ + c;
        attnout[o] = gm * d + pooled[o];
    }
}

// ---------------------------------------------------------------- z = attnout @ W2[seg], bf16 scatter to B-fragment order
// grid: B*50 blocks, 256 threads. Zb[b][ntile(16)][ks2(2)][lane(32)][e(16)] bf16; token k = K index (pad 50..63 zero)
__global__ __launch_bounds__(256) void k_zpack(const float* __restrict__ attnout, const float* __restrict__ W2,
                                               __bf16* __restrict__ Zb) {
    int id = blockIdx.x;
    int b = id / NTOK, t = id % NTOK;
    int lvl, iy, ix, li;
    tok_decode(t, lvl, iy, ix, li);
    int seg = li * 256;
    int oc = threadIdx.x;
    const float* ar  = attnout + (size_t)(b * NTOK + t) * C_;
    const float* w2c = W2 + (size_t)seg * 256 + oc;
    float acc = 0.0f;
    for (int k = 0; k < 256; ++k) acc += ar[k] * w2c[(size_t)k * 256];
    int ks2  = t >> 5;                       // K-step within the 64-wide gather block
    int r    = t & 31;
    int lane = (((r >> 3) & 1) << 4) | (oc & 15);  // lanes 0-15: K=0-7,16-23 ; lanes 16-31: K=8-15,24-31
    int e    = (r & 7) | ((r >> 4) << 3);
    int nt   = oc >> 4;
    Zb[(((((size_t)b * 16 + nt) * 2 + ks2) * 32 + lane) << 4) + e] = (__bf16)acc;
}

// ---------------------------------------------------------------- weight packing into bf16 B-fragments
// grid: 256 blocks (wsel*128 + ntile*8 + kstep), 512 threads (lane*16 + e)
__global__ __launch_bounds__(512) void k_wpack(const float* __restrict__ W1, const float* __restrict__ W2,
                                               __bf16* __restrict__ bw1, __bf16* __restrict__ bw2) {
    int id   = blockIdx.x;
    int wsel = id >> 7;
    int nt   = (id >> 3) & 15;
    int ks   = id & 7;
    int lane = threadIdx.x >> 4;
    int e    = threadIdx.x & 15;
    int K  = ks * 32 + ((e >> 3) << 4) + ((lane >> 4) << 3) + (e & 7);   // table order: e<8 K=kb+8h+e; e>=8 K=kb+16+8h+e-8
    int ch = nt * 16 + (lane & 15);
    float v = wsel ? W2[(size_t)(1024 + K) * 256 + ch] : W1[(size_t)K * 256 + ch];
    __bf16* dst = wsel ? bw2 : bw1;
    dst[((((size_t)nt * 8 + ks) * 32 + lane) << 4) + e] = (__bf16)v;
}

// ---------------------------------------------------------------- fold BN constants
__global__ __launch_bounds__(256) void k_fold(const float* b1, const float* s1, const float* o1,
                                              const float* m1, const float* v1,
                                              const float* b2, const float* s2, const float* o2,
                                              const float* m2, const float* v2,
                                              float* __restrict__ folds) {
    int c = threadIdx.x;
    float r1 = rsqrtf(v1[c] + EPS_) * s1[c];
    float r2 = rsqrtf(v2[c] + EPS_) * s2[c];
    folds[c]       = r1;                               // A1f
    folds[256 + c] = (b1[c] - m1[c]) * r1 + o1[c];     // B1f
    folds[512 + c] = r2;                               // A2f
    folds[768 + c] = (b2[c] - m2[c]) * r2 + o2[c];     // B2f (includes bias b2)
}

// ---------------------------------------------------------------- helper: load one 16x32 A-fragment from f32 rows
__device__ __forceinline__ v16bf load_a_frag(const float* xr, int kb, int lh) {
    float4 fa0 = *(const float4*)(xr + kb + lh * 8);
    float4 fa1 = *(const float4*)(xr + kb + lh * 8 + 4);
    float4 fb0 = *(const float4*)(xr + kb + 16 + lh * 8);
    float4 fb1 = *(const float4*)(xr + kb + 16 + lh * 8 + 4);
    v16bf a;
    a[0]  = (__bf16)fa0.x; a[1]  = (__bf16)fa0.y; a[2]  = (__bf16)fa0.z; a[3]  = (__bf16)fa0.w;
    a[4]  = (__bf16)fa1.x; a[5]  = (__bf16)fa1.y; a[6]  = (__bf16)fa1.z; a[7]  = (__bf16)fa1.w;
    a[8]  = (__bf16)fb0.x; a[9]  = (__bf16)fb0.y; a[10] = (__bf16)fb0.z; a[11] = (__bf16)fb0.w;
    a[12] = (__bf16)fb1.x; a[13] = (__bf16)fb1.y; a[14] = (__bf16)fb1.z; a[15] = (__bf16)fb1.w;
    return a;
}

// ---------------------------------------------------------------- fused main kernel
// grid: NPIX/128 blocks, 128 threads (4 waves); each wave owns 32 consecutive pixels (two 16-row tiles
// sharing every B fragment -> halves weight traffic from L2)
__global__ __launch_bounds__(128) void k_fused(const float* __restrict__ x,
                                               const __bf16* __restrict__ bw1, const __bf16* __restrict__ bw2,
                                               const __bf16* __restrict__ Zb,
                                               const float* __restrict__ folds, float* __restrict__ out) {
    __shared__ __bf16 ldsT[4][32][256];     // 64 KB, wave-private 16 KB slabs
    const int lane = threadIdx.x & 31;
    const int wv   = threadIdx.x >> 5;
    const int lh   = lane >> 4;             // lane half
    const int ln   = lane & 15;
    const int P0 = blockIdx.x * 128 + wv * 32;
    const int b  = P0 / PIXPB;
    const int rp = P0 % PIXPB;
    const int h  = rp / W_;
    const int w0 = rp % W_;                 // 32 | 192 -> whole tile sits in one row

    // ---- A1 fragments: x tiles (f32 -> bf16), WMMA 16-bit A layout ----
    v16bf a1A[8], a1B[8];
    {
        const float* xrA = x + (size_t)(P0 + ln) * C_;
        const float* xrB = x + (size_t)(P0 + 16 + ln) * C_;
#pragma unroll
        for (int ks = 0; ks < 8; ++ks) {
            a1A[ks] = load_a_frag(xrA, ks * 32, lh);
            a1B[ks] = load_a_frag(xrB, ks * 32, lh);
        }
    }

    // ---- bilinear gather fragments in bf16 A layout (K = token index, 2 K-steps of 32) ----
    v16bf gA[2], gB[2];
#pragma unroll
    for (int j = 0; j < 2; ++j) {
        v16bf a, bb;
#pragma unroll
        for (int e = 0; e < 16; ++e) {
            int k = j * 32 + afrag_k(lh, e);
            a[e]  = (__bf16)tokw(k, h, w0 + ln);
            bb[e] = (__bf16)tokw(k, h, w0 + 16 + ln);
        }
        gA[j] = a; gB[j] = bb;
    }

    // ---- GEMM1: t = relu((x@W1)*A1f + B1f) -> bf16 in wave-private LDS (natural [M][ch]) ----
    const v16bf* bw1p = (const v16bf*)bw1;
    for (int nt = 0; nt < 16; ++nt) {
        v8f accA = {0.f,0.f,0.f,0.f,0.f,0.f,0.f,0.f};
        v8f accB = {0.f,0.f,0.f,0.f,0.f,0.f,0.f,0.f};
#pragma unroll
        for (int ks = 0; ks < 8; ++ks) {
            v16bf bwf = bw1p[(nt * 8 + ks) * 32 + lane];
            accA = __builtin_amdgcn_wmma_f32_16x16x32_bf16(false, a1A[ks], false, bwf, (short)0, accA, false, false);
            accB = __builtin_amdgcn_wmma_f32_16x16x32_bf16(false, a1B[ks], false, bwf, (short)0, accB, false, false);
        }
        const int ch = nt * 16 + ln;
        const float sA = folds[ch];
        const float sB = folds[256 + ch];
#pragma unroll
        for (int r = 0; r < 8; ++r) {        // D layout: lanes0-15 M=r, lanes16-31 M=8+r
            ldsT[wv][lh * 8 + r][ch]      = (__bf16)fmaxf(accA[r] * sA + sB, 0.0f);
            ldsT[wv][16 + lh * 8 + r][ch] = (__bf16)fmaxf(accB[r] * sA + sB, 0.0f);
        }
    }

    // ---- reload t as A2 fragments ----
    v16bf a2A[8], a2B[8];
#pragma unroll
    for (int ks = 0; ks < 8; ++ks) {
        const int kb = ks * 32;
        v8bf loA = *(const v8bf*)&ldsT[wv][ln][kb + lh * 8];
        v8bf hiA = *(const v8bf*)&ldsT[wv][ln][kb + 16 + lh * 8];
        v8bf loB = *(const v8bf*)&ldsT[wv][16 + ln][kb + lh * 8];
        v8bf hiB = *(const v8bf*)&ldsT[wv][16 + ln][kb + 16 + lh * 8];
        v16bf a, bb;
#pragma unroll
        for (int e = 0; e < 8; ++e) { a[e] = loA[e]; a[8 + e] = hiA[e]; bb[e] = loB[e]; bb[8 + e] = hiB[e]; }
        a2A[ks] = a; a2B[ks] = bb;
    }

    // ---- GEMM2 (K=256 weights + K=64 bilinear gather), fold BN2, relu, store ----
    const v16bf* bw2p = (const v16bf*)bw2;
    const v16bf* zbp  = (const v16bf*)Zb;
    for (int nt = 0; nt < 16; ++nt) {
        v8f accA = {0.f,0.f,0.f,0.f,0.f,0.f,0.f,0.f};
        v8f accB = {0.f,0.f,0.f,0.f,0.f,0.f,0.f,0.f};
#pragma unroll
        for (int ks = 0; ks < 8; ++ks) {
            v16bf bwf = bw2p[(nt * 8 + ks) * 32 + lane];
            accA = __builtin_amdgcn_wmma_f32_16x16x32_bf16(false, a2A[ks], false, bwf, (short)0, accA, false, false);
            accB = __builtin_amdgcn_wmma_f32_16x16x32_bf16(false, a2B[ks], false, bwf, (short)0, accB, false, false);
        }
#pragma unroll
        for (int j = 0; j < 2; ++j) {
            v16bf zf = zbp[(((size_t)b * 16 + nt) * 2 + j) * 32 + lane];
            accA = __builtin_amdgcn_wmma_f32_16x16x32_bf16(false, gA[j], false, zf, (short)0, accA, false, false);
            accB = __builtin_amdgcn_wmma_f32_16x16x32_bf16(false, gB[j], false, zf, (short)0, accB, false, false);
        }
        const int ch = nt * 16 + ln;
        const float sA = folds[512 + ch];
        const float sB = folds[768 + ch];
#pragma unroll
        for (int r = 0; r < 8; ++r) {
            out[(size_t)(P0 + lh * 8 + r) * C_ + ch]      = fmaxf(accA[r] * sA + sB, 0.0f);
            out[(size_t)(P0 + 16 + lh * 8 + r) * C_ + ch] = fmaxf(accB[r] * sA + sB, 0.0f);
        }
    }
}

// ================================================================ host launch
extern "C" void kernel_launch(void* const* d_in, const int* in_sizes, int n_in,
                              void* d_out, int out_size, void* d_ws, size_t ws_size,
                              hipStream_t stream) {
    (void)in_sizes; (void)n_in; (void)out_size; (void)ws_size;
    const float* x = (const float*)d_in[0];

    AttnPtrs ap;
    for (int a = 0; a < 4; ++a) {
        int base = 1 + 9 * a;   // wf, bf, wg, bg, wh, bh, wo, bo, gamma
        ap.wf[a]    = (const float*)d_in[base + 0];
        ap.bf[a]    = (const float*)d_in[base + 1];
        ap.wg[a]    = (const float*)d_in[base + 2];
        ap.bg[a]    = (const float*)d_in[base + 3];
        ap.wh[a]    = (const float*)d_in[base + 4];
        ap.bh[a]    = (const float*)d_in[base + 5];
        ap.wo[a]    = (const float*)d_in[base + 6];
        ap.bo[a]    = (const float*)d_in[base + 7];
        ap.gamma[a] = (const float*)d_in[base + 8];
    }
    // proj1: w,b,scale,offset,mean,var at 37..42 ; proj2 at 43..48
    const float* W1  = (const float*)d_in[37];
    const float* b1  = (const float*)d_in[38];
    const float* s1  = (const float*)d_in[39];
    const float* o1  = (const float*)d_in[40];
    const float* m1  = (const float*)d_in[41];
    const float* v1  = (const float*)d_in[42];
    const float* W2  = (const float*)d_in[43];
    const float* b2  = (const float*)d_in[44];
    const float* s2  = (const float*)d_in[45];
    const float* o2  = (const float*)d_in[46];
    const float* m2  = (const float*)d_in[47];
    const float* v2  = (const float*)d_in[48];
    float* outp = (float*)d_out;

    char* ws = (char*)d_ws;
    float*  part6   = (float*) (ws + 0);         // 4*36*8*256 f    = 1179648 B
    float*  pooled  = (float*) (ws + 1179648);   // 4*50*256 f      =  204800 B
    float*  fgh     = (float*) (ws + 1384448);   // 4*50*192 f      =  153600 B
    float*  attnout = (float*) (ws + 1538048);   // 4*50*256 f      =  204800 B
    float*  folds   = (float*) (ws + 1742848);   // 4*256 f         =    4096 B
    __bf16* bw1     = (__bf16*)(ws + 1746944);   // 16*8*32*16 bf16 =  131072 B
    __bf16* bw2     = (__bf16*)(ws + 1878016);   // 131072 B
    __bf16* Zb      = (__bf16*)(ws + 2009088);   // 4*16*2*32*16 bf16 = 131072 B

    const int ZN = 131072 / 4;                   // zero Zb (pads must be 0)
    k_init    <<<(ZN + 255) / 256, 256, 0, stream>>>((float*)Zb, ZN);
    k_pool6   <<<B_ * 36 * 8, 256, 0, stream>>>(x, part6);
    k_pool_all<<<B_ * NTOK,   256, 0, stream>>>(part6, pooled);
    k_fold    <<<1, 256, 0, stream>>>(b1, s1, o1, m1, v1, b2, s2, o2, m2, v2, folds);
    k_qkv     <<<B_ * NTOK, 192, 0, stream>>>(pooled, fgh, ap);
    k_attn    <<<B_ * 4, 256, 0, stream>>>(pooled, fgh, attnout, ap);
    k_zpack   <<<B_ * NTOK, 256, 0, stream>>>(attnout, W2, Zb);
    k_wpack   <<<256, 512, 0, stream>>>(W1, W2, bw1, bw2);
    k_fused   <<<NPIX / 128, 128, 0, stream>>>(x, bw1, bw2, Zb, folds, outp);
}
